// Network_38388417691728
// MI455X (gfx1250) — matrix operand order repaired and checked by reading, not compiled, not run on hardware
//
#include <hip/hip_runtime.h>
#include <math.h>

typedef __attribute__((ext_vector_type(16))) _Float16 v16h;
typedef __attribute__((ext_vector_type(8)))  float    v8f;

#define XSTR   296      // LDS row stride in halves (16B-aligned: 296*2=592=16*37)
#define NRAYS  1024
#define SC     64       // coarse samples per ray
#define SF     192      // fine samples per ray
#define NIMP   128
#define BGCOL  1.0f

// ---------------- output layout (floats) ----------------
#define OUT_FRGB  0
#define OUT_FW    (OUT_FRGB + NRAYS*3)        // 3072
#define OUT_FACC  (OUT_FW   + NRAYS*SF)       // 199680
#define OUT_CRGB  (OUT_FACC + NRAYS)          // 200704
#define OUT_CW    (OUT_CRGB + NRAYS*3)        // 203776
#define OUT_CACC  (OUT_CW   + NRAYS*SC)       // 269312

// ---------------- packed weight layout per net (halves) ----------------
#define PW_XYZ0   0                       // 64x256
#define PW_XYZ(i) (16384 + (i)*65536)     // 7 x 256x256
#define PW_FEAT   (16384 + 7*65536)       // 256x256  = 475136
#define PW_DIR    (PW_FEAT + 65536)       // 288x128  = 540672
#define PW_NET    (PW_DIR + 288*128)      // 577536 halves per net

struct NetParams {
  const _Float16* wx0;
  const _Float16* wx[7];
  const _Float16* wfeat;
  const _Float16* wdir;
  const float* bx[8];
  const float* bfeat;
  const float* bdir;
  const float* wAlpha; const float* bAlpha;
  const float* wRgb;   const float* bRgb;
};

// =====================================================================
// Weight packing: f32 (Ksrc x N row-major) -> f16 WMMA B-fragment layout.
// For tile (kt,nt): lane holds column nt*16+(lane&15); its 16 halves are
// K = kt*32 + (i>>3)*16 + (lane>>4)*8 + (i&7)  (i=0..15), stored contiguously (32B/lane).
// =====================================================================
__global__ void pack_w_kernel(const float* __restrict__ src, _Float16* __restrict__ dst,
                              int Ksrc, int Kpad, int N) {
  int idx = blockIdx.x * blockDim.x + threadIdx.x;
  int total = Kpad * N;
  if (idx >= total) return;
  int nnt  = N >> 4;
  int i    = idx & 15;
  int lane = (idx >> 4) & 31;
  int tile = idx >> 9;              // kt*nnt + nt
  int kt = tile / nnt, nt = tile - kt * nnt;
  int g   = lane >> 4;
  int col = nt * 16 + (lane & 15);
  int k   = kt * 32 + (i >> 3) * 16 + g * 8 + (i & 7);
  float v = (k < Ksrc) ? src[k * N + col] : 0.0f;
  dst[idx] = (_Float16)v;
}

// =====================================================================
// Wave-cooperative GEMM layer: Xout(32xN) = act(Xin(32xK) @ Wp + bias)
// 4 waves: wave>>1 selects 16-row stripe, wave&1 selects column half.
// A from LDS (16-bit A layout: lane r16 row, g selects K-half; two
// contiguous 8-half chunks). B from packed global (one v16h per lane).
// =====================================================================
__device__ __forceinline__ void gemm_layer(const _Float16* Xin, _Float16* Xout,
                                           const _Float16* __restrict__ Wp,
                                           const float* __restrict__ bias,
                                           int K, int N, bool relu) {
  const int t    = threadIdx.x;
  const int lane = t & 31;
  const int wv   = t >> 5;
  const int rowBase = (wv >> 1) * 16;
  const int r16 = lane & 15;
  const int g   = lane >> 4;
  const int nnt = N >> 4;
  const int nkt = K >> 5;
  const int ntBeg = (wv & 1) * (nnt >> 1);
  const int ntEnd = ntBeg + (nnt >> 1);
  const v16h* __restrict__ Bp = (const v16h*)Wp;

  for (int nt = ntBeg; nt < ntEnd; ++nt) {
    v8f acc = {};
    for (int kt = 0; kt < nkt; ++kt) {
      v16h a;
      const _Float16* ar = Xin + (rowBase + r16) * XSTR + kt * 32 + g * 8;
      #pragma unroll
      for (int i = 0; i < 8; ++i) { a[i] = ar[i]; a[8 + i] = ar[16 + i]; }
      v16h b = Bp[(kt * nnt + nt) * 32 + lane];
      acc = __builtin_amdgcn_wmma_f32_16x16x32_f16(false, a, false, b,
                                                   (short)0, acc, false, false);
    }
    const int col = nt * 16 + r16;
    const float bb = bias[col];
    #pragma unroll
    for (int r = 0; r < 8; ++r) {
      float v = acc[r] + bb;
      if (relu) v = v > 0.f ? v : 0.f;
      Xout[(rowBase + r + 8 * g) * XSTR + col] = (_Float16)v;
    }
  }
  __syncthreads();
}

// =====================================================================
// Full MLP for a tile of 32 samples of one ray.
// Builds posenc input in LDS, runs xyz trunk + heads, writes rawAlpha/rgb.
// =====================================================================
__launch_bounds__(128)
__global__ void mlp_kernel(const float* __restrict__ rays_o, const float* __restrict__ rays_d,
                           const float* __restrict__ nearp, const float* __restrict__ farp,
                           const float* __restrict__ zbuf, int S,
                           NetParams P,
                           float* __restrict__ alphaOut, float* __restrict__ rgbOut) {
  const int tilesPerRay = S >> 5;
  const int ray = blockIdx.x / tilesPerRay;
  const int sub = blockIdx.x - ray * tilesPerRay;
  const int t = threadIdx.x;

  __shared__ _Float16 X0[32 * XSTR];
  __shared__ _Float16 X1[32 * XSTR];
  __shared__ float dirE[32];

  const float o0 = rays_o[ray*3+0], o1 = rays_o[ray*3+1], o2 = rays_o[ray*3+2];
  const float d0 = rays_d[ray*3+0], d1 = rays_d[ray*3+1], d2 = rays_d[ray*3+2];
  const float nr = nearp[ray], fr = farp[ray];
  const float dn = sqrtf(d0*d0 + d1*d1 + d2*d2);

  // view-direction positional encoding (27 channels), once per tile
  if (t < 32) {
    float v = 0.f;
    if (t < 27) {
      const float u0 = d0/dn, u1 = d1/dn, u2 = d2/dn;
      if (t < 3) v = (t==0)?u0:((t==1)?u1:u2);
      else {
        int l = (t-3)/6, m = (t-3)%6;
        int ax = (m < 3) ? m : m-3;
        float base = (ax==0)?u0:((ax==1)?u1:u2);
        float w = base * exp2f((float)l);
        v = (m < 3) ? sinf(w) : cosf(w);
      }
    }
    dirE[t] = v;
  }

  // xyz positional encoding into X0: 32 rows x 64 cols (col 63 = zero pad)
  for (int idx = t; idx < 32 * 64; idx += 128) {
    int r = idx >> 6, c = idx & 63;
    int j = sub * 32 + r;
    float tt = (float)j / (float)(S - 1);
    float z  = zbuf ? zbuf[ray * S + j] : (nr * (1.f - tt) + fr * tt);
    float val = 0.f;
    if (c < 3) {
      val = ((c==0)?o0:((c==1)?o1:o2)) + ((c==0)?d0:((c==1)?d1:d2)) * z;
    } else if (c < 63) {
      int l = (c-3)/6, m = (c-3)%6;
      int ax = (m < 3) ? m : m-3;
      float p = ((ax==0)?o0:((ax==1)?o1:o2)) + ((ax==0)?d0:((ax==1)?d1:d2)) * z;
      float w = p * exp2f((float)l);
      val = (m < 3) ? sinf(w) : cosf(w);
    }
    X0[r * XSTR + c] = (_Float16)val;
  }
  __syncthreads();

  // xyz trunk
  gemm_layer(X0, X1, P.wx0, P.bx[0], 64, 256, true);
  _Float16* in  = X1;
  _Float16* out = X0;
  for (int l = 0; l < 7; ++l) {
    gemm_layer(in, out, P.wx[l], P.bx[l+1], 256, 256, true);
    _Float16* tmp = in; in = out; out = tmp;
  }
  // 'in' now holds x (32 x 256)

  // alpha head (raw, relu applied in render)
  if (t < 32) {
    float a = P.bAlpha[0];
    for (int k = 0; k < 256; ++k) a += (float)in[t * XSTR + k] * P.wAlpha[k];
    alphaOut[ray * S + sub * 32 + t] = a;
  }

  // feature head (no activation) -> 'out' cols 0..255
  gemm_layer(in, out, P.wfeat, P.bfeat, 256, 256, false);

  // append dir encoding, cols 256..287 (pad with zeros)
  for (int idx = t; idx < 32 * 32; idx += 128) {
    int r = idx >> 5, c = idx & 31;
    out[r * XSTR + 256 + c] = (_Float16)((c < 27) ? dirE[c] : 0.f);
  }
  __syncthreads();
  { _Float16* tmp = in; in = out; out = tmp; }

  // dir layer: 288 -> 128, relu
  gemm_layer(in, out, P.wdir, P.bdir, 288, 128, true);

  // rgb head: 128 -> 3, sigmoid (96 threads: 32 samples x 3 channels)
  if (t < 96) {
    int s = t / 3, c = t - 3 * s;
    float v = P.bRgb[c];
    for (int k = 0; k < 128; ++k) v += (float)out[s * XSTR + k] * P.wRgb[k * 3 + c];
    v = 1.f / (1.f + expf(-v));
    rgbOut[(ray * S + sub * 32 + s) * 3 + c] = v;
  }
}

// =====================================================================
// Coarse volume render + hierarchical PDF sampling (1 thread per ray)
// =====================================================================
__global__ void render_coarse_kernel(const float* __restrict__ rays_o, const float* __restrict__ rays_d,
                                     const float* __restrict__ nearp, const float* __restrict__ farp,
                                     const float* __restrict__ alphaRaw, const float* __restrict__ rgbS,
                                     float* __restrict__ zfine, float* __restrict__ out) {
  int ray = blockIdx.x * blockDim.x + threadIdx.x;
  if (ray >= NRAYS) return;
  const float d0 = rays_d[ray*3+0], d1 = rays_d[ray*3+1], d2 = rays_d[ray*3+2];
  const float dn = sqrtf(d0*d0 + d1*d1 + d2*d2);
  const float nr = nearp[ray], fr = farp[ray];

  float z[SC], w[SC];
  for (int i = 0; i < SC; ++i) {
    float tt = (float)i / (float)(SC - 1);
    z[i] = nr * (1.f - tt) + fr * tt;
  }
  float T = 1.f, acc = 0.f, r0 = 0.f, r1 = 0.f, r2 = 0.f;
  for (int i = 0; i < SC; ++i) {
    float dist = ((i < SC-1) ? (z[i+1] - z[i]) : 1e10f) * dn;
    float raw = alphaRaw[ray * SC + i]; raw = raw > 0.f ? raw : 0.f;
    float a = 1.f - expf(-raw * dist);
    float wi = a * T;
    w[i] = wi; acc += wi;
    r0 += wi * rgbS[(ray*SC+i)*3+0];
    r1 += wi * rgbS[(ray*SC+i)*3+1];
    r2 += wi * rgbS[(ray*SC+i)*3+2];
    T *= (1.f - a + 1e-10f);
  }
  out[OUT_CRGB + ray*3+0] = r0 + BGCOL * (1.f - acc);
  out[OUT_CRGB + ray*3+1] = r1 + BGCOL * (1.f - acc);
  out[OUT_CRGB + ray*3+2] = r2 + BGCOL * (1.f - acc);
  for (int i = 0; i < SC; ++i) out[OUT_CW + ray*SC + i] = w[i];
  out[OUT_CACC + ray] = acc;

  // sample_pdf: bins = z_mid (63), weights = w[1..62]
  float bins[SC-1], cdf[SC-1];
  for (int i = 0; i < SC-1; ++i) bins[i] = 0.5f * (z[i] + z[i+1]);
  float sum = 0.f;
  for (int i = 0; i < SC-2; ++i) sum += w[i+1] + 1e-5f;
  cdf[0] = 0.f;
  for (int i = 0; i < SC-2; ++i) cdf[i+1] = cdf[i] + (w[i+1] + 1e-5f) / sum;

  float zs[NIMP];
  for (int i = 0; i < NIMP; ++i) {
    float u = (float)i / (float)(NIMP - 1);
    // searchsorted(cdf, u, side='right') over 63 entries
    int lo = 0, hi = SC-1;
    while (lo < hi) { int mid = (lo + hi) >> 1; if (cdf[mid] <= u) lo = mid + 1; else hi = mid; }
    int below = lo - 1; if (below < 0) below = 0; if (below > SC-2) below = SC-2;
    int above = lo;     if (above > SC-2) above = SC-2;
    float c0 = cdf[below], c1 = cdf[above];
    float den = c1 - c0; if (den < 1e-5f) den = 1.f;
    float tt = (u - c0) / den;
    zs[i] = bins[below] + tt * (bins[above] - bins[below]);
  }
  // merge sorted z (64) and zs (128) -> fine z (192)
  int a = 0, b = 0;
  float* dst = zfine + ray * SF;
  for (int k = 0; k < SF; ++k) {
    float za = (a < SC)   ? z[a]  : 3.4e38f;
    float zb = (b < NIMP) ? zs[b] : 3.4e38f;
    if (za <= zb) { dst[k] = za; ++a; } else { dst[k] = zb; ++b; }
  }
}

// =====================================================================
// Fine volume render (1 thread per ray)
// =====================================================================
__global__ void render_fine_kernel(const float* __restrict__ rays_d,
                                   const float* __restrict__ zfine,
                                   const float* __restrict__ alphaRaw, const float* __restrict__ rgbS,
                                   float* __restrict__ out) {
  int ray = blockIdx.x * blockDim.x + threadIdx.x;
  if (ray >= NRAYS) return;
  const float d0 = rays_d[ray*3+0], d1 = rays_d[ray*3+1], d2 = rays_d[ray*3+2];
  const float dn = sqrtf(d0*d0 + d1*d1 + d2*d2);
  const float* zf = zfine + ray * SF;

  float T = 1.f, acc = 0.f, r0 = 0.f, r1 = 0.f, r2 = 0.f;
  for (int i = 0; i < SF; ++i) {
    float dist = ((i < SF-1) ? (zf[i+1] - zf[i]) : 1e10f) * dn;
    float raw = alphaRaw[ray * SF + i]; raw = raw > 0.f ? raw : 0.f;
    float a = 1.f - expf(-raw * dist);
    float wi = a * T;
    out[OUT_FW + ray*SF + i] = wi;
    acc += wi;
    r0 += wi * rgbS[(ray*SF+i)*3+0];
    r1 += wi * rgbS[(ray*SF+i)*3+1];
    r2 += wi * rgbS[(ray*SF+i)*3+2];
    T *= (1.f - a + 1e-10f);
  }
  out[OUT_FRGB + ray*3+0] = r0 + BGCOL * (1.f - acc);
  out[OUT_FRGB + ray*3+1] = r1 + BGCOL * (1.f - acc);
  out[OUT_FRGB + ray*3+2] = r2 + BGCOL * (1.f - acc);
  out[OUT_FACC + ray] = acc;
}

// =====================================================================
// Host-side launch
// Input order (JAX pytree flatten, dict keys sorted):
//   0:rays_o 1:rays_d 2:near 3:far
//   coarse @8 / fine @32, each: alphaW,alphab,dirW,dirb,featW,featb,
//   rgbW,rgbb, xyz0W,xyz0b ... xyz7W,xyz7b
// =====================================================================
extern "C" void kernel_launch(void* const* d_in, const int* in_sizes, int n_in,
                              void* d_out, int out_size, void* d_ws, size_t ws_size,
                              hipStream_t stream) {
  const float* rays_o = (const float*)d_in[0];
  const float* rays_d = (const float*)d_in[1];
  const float* nearp  = (const float*)d_in[2];
  const float* farp   = (const float*)d_in[3];
  float* out = (float*)d_out;

  // workspace carve-up
  _Float16* wpack = (_Float16*)d_ws;
  float* fbase  = (float*)((char*)d_ws + (size_t)2 * PW_NET * sizeof(_Float16));
  float* zfine  = fbase;                   // 1024*192
  float* alphaC = zfine  + NRAYS * SF;     // 65536
  float* rgbC   = alphaC + NRAYS * SC;     // 196608
  float* alphaF = rgbC   + NRAYS * SC * 3; // 196608
  float* rgbF   = alphaF + NRAYS * SF;     // 589824

  auto packW = [&](const void* src, _Float16* dst, int Ksrc, int Kpad, int N) {
    int total = Kpad * N;
    pack_w_kernel<<<dim3((total + 255) / 256), dim3(256), 0, stream>>>(
        (const float*)src, dst, Ksrc, Kpad, N);
  };

  NetParams NP[2];
  for (int net = 0; net < 2; ++net) {
    int base = 4 + net * 24;
    _Float16* W = wpack + (size_t)net * PW_NET;
    // pack weights into WMMA fragment layout
    packW(d_in[base + 12], W + PW_XYZ0, 63, 64, 256);
    for (int i = 1; i < 8; ++i)
      packW(d_in[base + 12 + 2*i], W + PW_XYZ(i-1), 256, 256, 256);
    packW(d_in[base + 8], W + PW_FEAT, 256, 256, 256);
    packW(d_in[base + 6], W + PW_DIR, 283, 288, 128);

    NetParams& P = NP[net];
    P.wx0 = W + PW_XYZ0;
    for (int i = 0; i < 7; ++i) P.wx[i] = W + PW_XYZ(i);
    P.wfeat = W + PW_FEAT;
    P.wdir  = W + PW_DIR;
    for (int i = 0; i < 8; ++i) P.bx[i] = (const float*)d_in[base + 13 + 2*i];
    P.bfeat  = (const float*)d_in[base + 9];
    P.bdir   = (const float*)d_in[base + 7];
    P.wAlpha = (const float*)d_in[base + 4];
    P.bAlpha = (const float*)d_in[base + 5];
    P.wRgb   = (const float*)d_in[base + 10];
    P.bRgb   = (const float*)d_in[base + 11];
  }

  // coarse MLP: 1024 rays x 2 tiles of 32 samples
  mlp_kernel<<<dim3(NRAYS * (SC/32)), dim3(128), 0, stream>>>(
      rays_o, rays_d, nearp, farp, (const float*)nullptr, SC, NP[0], alphaC, rgbC);

  // coarse render + hierarchical sampling
  render_coarse_kernel<<<dim3((NRAYS + 255) / 256), dim3(256), 0, stream>>>(
      rays_o, rays_d, nearp, farp, alphaC, rgbC, zfine, out);

  // fine MLP: 1024 rays x 6 tiles
  mlp_kernel<<<dim3(NRAYS * (SF/32)), dim3(128), 0, stream>>>(
      rays_o, rays_d, nearp, farp, zfine, SF, NP[1], alphaF, rgbF);

  // fine render
  render_fine_kernel<<<dim3((NRAYS + 255) / 256), dim3(256), 0, stream>>>(
      rays_d, zfine, alphaF, rgbF, out);
}
